// MinGRU_32873679683590
// MI455X (gfx1250) — matrix operand order repaired and checked by reading, not compiled, not run on hardware
//
#include <hip/hip_runtime.h>
#include <hip/hip_bf16.h>

// Problem constants (fixed by the reference): B=8, T=4096, H=1024, 2 layers.
#define BB   8
#define TT   4096
#define HH   1024
#define MDIM (BB * TT)       // 32768 rows of the GEMM
#define KDIM (HH)            // 1024
#define NDIM (3 * HH)        // 3072
#define NC   64              // scan chunks along T
#define CL   (TT / NC)       // 64 steps per chunk

#define PP 20                // LDS row pitch (dwords): 16 K-pair dwords + pad;
                             // keeps every 4-dword group 16B-aligned for B128
                             // and 20*l mod 64 distinct across a 16-lane group.

typedef __attribute__((ext_vector_type(16))) __bf16 v16bf;
typedef __attribute__((ext_vector_type(8)))  float  v8f;

union Frag {
  unsigned int u[8];
  uint4 q[2];
  v16bf v;
};

// Split f32 into bf16 hi/lo parts (truncating split: a = hi + lo + O(2^-16)*a).
__device__ __forceinline__ void bsplit(float a, unsigned int& hi,
                                       unsigned int& lo) {
  const unsigned int u = __builtin_bit_cast(unsigned int, a);
  const float fh = __builtin_bit_cast(float, u & 0xFFFF0000u);
  const float r  = a - fh;  // exact
  hi = u >> 16;
  lo = __builtin_bit_cast(unsigned int, r) >> 16;
}

// ---------------------------------------------------------------------------
// GEMM: Z = A(f32) x W(f32) via split-bf16 WMMA:
//   D += Ah*Bh + Ah*Bl + Al*Bh   (V_WMMA_F32_16X16X32_BF16, f32 accumulate)
// Block tile 128(M) x 64(N), 8 waves (4x2), wave tile 32x32 = 2x2 WMMA tiles,
// K staged in 32-wide double-buffered LDS chunks. LDS is element-major
// ([m][kpair] / [n][kpair]) so each lane's fragment is two DS_LOAD_B128s.
// ---------------------------------------------------------------------------
__global__ __launch_bounds__(256) void mingru_gemm_splitbf16(
    const float* __restrict__ A, const float* __restrict__ W,
    float* __restrict__ Z) {
  // dword [e][kp] packs bf16(K=2kp) | bf16(K=2kp+1)<<16
  __shared__ unsigned int AsH[2][128][PP];
  __shared__ unsigned int AsL[2][128][PP];
  __shared__ unsigned int BsH[2][64][PP];
  __shared__ unsigned int BsL[2][64][PP];

  const int tid  = threadIdx.x;
  const int m0   = blockIdx.x * 128;
  const int n0   = blockIdx.y * 64;
  const int lane = tid & 31;
  const int wave = tid >> 5;
  const int wm   = (wave & 3) << 5;  // wave offset along M (0,32,64,96)
  const int wn   = (wave >> 2) << 5; // wave offset along N (0,32)
  const int half = lane >> 4;        // K-half selector (A/B 16-bit layouts)
  const int l    = lane & 15;

  v8f acc[2][2];
#pragma unroll
  for (int i = 0; i < 2; ++i)
#pragma unroll
    for (int j = 0; j < 2; ++j) acc[i][j] = (v8f)0.0f;

  // Stage one 32-wide K chunk into LDS buffer `buf` (f32 -> bf16 hi/lo split).
  auto stage = [&](int k0, int buf) {
    // A tile 128x32 -> AsH/AsL[m][kp]; K-pairs are consecutive dwords in a row.
#pragma unroll
    for (int i = tid; i < 128 * 8; i += 256) {
      const int m  = i >> 3;
      const int kq = i & 7;  // group of 4 consecutive K -> kpairs 2kq, 2kq+1
      const float4 v =
          *(const float4*)(A + (size_t)(m0 + m) * KDIM + k0 + kq * 4);
      unsigned int h0, l0, h1, l1, h2, l2, h3, l3;
      bsplit(v.x, h0, l0); bsplit(v.y, h1, l1);
      bsplit(v.z, h2, l2); bsplit(v.w, h3, l3);
      uint2 th, tl;
      th.x = h0 | (h1 << 16); th.y = h2 | (h3 << 16);
      tl.x = l0 | (l1 << 16); tl.y = l2 | (l3 << 16);
      *(uint2*)&AsH[buf][m][2 * kq] = th;
      *(uint2*)&AsL[buf][m][2 * kq] = tl;
    }
    // B tile 32x64 -> BsH/BsL[n][kp]: each thread does one (kpair, 4-wide N).
    {
      const int kp = tid >> 4;         // 0..15
      const int nq = (tid & 15) << 2;  // 0..60
      const float4 r0 =
          *(const float4*)(W + (size_t)(k0 + 2 * kp + 0) * NDIM + n0 + nq);
      const float4 r1 =
          *(const float4*)(W + (size_t)(k0 + 2 * kp + 1) * NDIM + n0 + nq);
      const float e0[4] = {r0.x, r0.y, r0.z, r0.w};
      const float e1[4] = {r1.x, r1.y, r1.z, r1.w};
#pragma unroll
      for (int j = 0; j < 4; ++j) {
        unsigned int ha, la, hb, lb;
        bsplit(e0[j], ha, la);
        bsplit(e1[j], hb, lb);
        BsH[buf][nq + j][kp] = ha | (hb << 16);
        BsL[buf][nq + j][kp] = la | (lb << 16);
      }
    }
  };

  const int nChunk = KDIM / 32;  // 32
  stage(0, 0);

  for (int c = 0; c < nChunk; ++c) {
    const int p = c & 1;
    __syncthreads();  // buffer p stores visible; buffer 1-p free to overwrite
    if (c + 1 < nChunk) stage((c + 1) * 32, 1 - p);

    // Fragment loads from buffer p — two DS_LOAD_B128 per fragment.
    // A 16x32 bf16 layout: all lanes M=l; kp runs 4*half+0..3 and 8+4*half+0..3.
    Frag ah[2], al[2], bh[2], bl[2];
#pragma unroll
    for (int tm = 0; tm < 2; ++tm) {
      const int mIdx = wm + tm * 16 + l;
      ah[tm].q[0] = *(const uint4*)&AsH[p][mIdx][4 * half];
      ah[tm].q[1] = *(const uint4*)&AsH[p][mIdx][8 + 4 * half];
      al[tm].q[0] = *(const uint4*)&AsL[p][mIdx][4 * half];
      al[tm].q[1] = *(const uint4*)&AsL[p][mIdx][8 + 4 * half];
    }
    // B 32x16 bf16 layout: N=l; kp runs 4*half+0..3 and 8+4*half+0..3.
#pragma unroll
    for (int tn = 0; tn < 2; ++tn) {
      const int nIdx = wn + tn * 16 + l;
      bh[tn].q[0] = *(const uint4*)&BsH[p][nIdx][4 * half];
      bh[tn].q[1] = *(const uint4*)&BsH[p][nIdx][8 + 4 * half];
      bl[tn].q[0] = *(const uint4*)&BsL[p][nIdx][4 * half];
      bl[tn].q[1] = *(const uint4*)&BsL[p][nIdx][8 + 4 * half];
    }

#pragma unroll
    for (int tm = 0; tm < 2; ++tm)
#pragma unroll
      for (int tn = 0; tn < 2; ++tn) {
        acc[tm][tn] = __builtin_amdgcn_wmma_f32_16x16x32_bf16(
            false, ah[tm].v, false, bh[tn].v, (short)0, acc[tm][tn],
            false, false);
        acc[tm][tn] = __builtin_amdgcn_wmma_f32_16x16x32_bf16(
            false, ah[tm].v, false, bl[tn].v, (short)0, acc[tm][tn],
            false, false);
        acc[tm][tn] = __builtin_amdgcn_wmma_f32_16x16x32_bf16(
            false, al[tm].v, false, bh[tn].v, (short)0, acc[tm][tn],
            false, false);
      }
  }

  // C/D f32 layout: VGPR r -> M = r (lanes 0-15) or r+8 (lanes 16-31), N = l.
#pragma unroll
  for (int tm = 0; tm < 2; ++tm)
#pragma unroll
    for (int tn = 0; tn < 2; ++tn)
#pragma unroll
      for (int r = 0; r < 8; ++r) {
        const int row = m0 + wm + tm * 16 + half * 8 + r;
        const int col = n0 + wn + tn * 16 + l;
        Z[(size_t)row * NDIM + col] = acc[tm][tn][r];
      }
}

// ---------------------------------------------------------------------------
// Scan: h_t = c_t * h_{t-1} + v_t, c_t = sigmoid(-gate), v_t = sigmoid(gate)*g(hidden)
// Chunked parallel scan over T (NC chunks of CL), then highway gate.
// ---------------------------------------------------------------------------
__device__ __forceinline__ float sigf(float x) {
  return 1.0f / (1.0f + __expf(-x));
}
__device__ __forceinline__ float gfun(float x) {
  return x >= 0.0f ? x + 0.5f : sigf(x);
}

// Pass 1: per (b, ch, chunk) local scan with zero init -> (prod, partial).
__global__ __launch_bounds__(256) void mingru_scan1(
    const float* __restrict__ Z, float* __restrict__ cp,
    float* __restrict__ cs) {
  const int ch    = ((blockIdx.x & 3) << 8) + threadIdx.x;    // 0..1023
  const int chunk = (blockIdx.x >> 2) & (NC - 1);             // 0..63
  const int b     = blockIdx.x >> 8;                          // 0..7
  float s = 0.0f, p = 1.0f;
  const int tbeg = chunk * CL;
#pragma unroll 4
  for (int t = tbeg; t < tbeg + CL; ++t) {
    const size_t base = (size_t)(b * TT + t) * NDIM;
    const float gate = Z[base + HH + ch];
    const float hid  = Z[base + ch];
    const float c = sigf(-gate);
    const float v = sigf(gate) * gfun(hid);
    s = fmaf(c, s, v);
    p *= c;
  }
  const size_t cidx = (size_t)(b * NC + chunk) * HH + ch;
  cp[cidx] = p;
  cs[cidx] = s;
}

// Pass 2: sequential carry combine across the NC chunks (tiny).
__global__ __launch_bounds__(256) void mingru_scan2(
    const float* __restrict__ cp, const float* __restrict__ cs,
    float* __restrict__ carry) {
  const int idx = blockIdx.x * 256 + threadIdx.x;  // 0..B*H-1
  const int b  = idx >> 10;
  const int ch = idx & (HH - 1);
  float s = 0.0f;
#pragma unroll 8
  for (int k = 0; k < NC; ++k) {
    const size_t cidx = (size_t)(b * NC + k) * HH + ch;
    carry[cidx] = s;                 // exclusive carry-in for chunk k
    s = fmaf(cp[cidx], s, cs[cidx]);
  }
}

// Pass 3: re-scan each chunk with its carry-in, fuse highway gate, write out.
__global__ __launch_bounds__(256) void mingru_scan3(
    const float* __restrict__ Z, const float* __restrict__ carry,
    const float* __restrict__ Hin, float* __restrict__ Hout) {
  const int ch    = ((blockIdx.x & 3) << 8) + threadIdx.x;
  const int chunk = (blockIdx.x >> 2) & (NC - 1);
  const int b     = blockIdx.x >> 8;
  float s = carry[(size_t)(b * NC + chunk) * HH + ch];
  const int tbeg = chunk * CL;
#pragma unroll 4
  for (int t = tbeg; t < tbeg + CL; ++t) {
    const size_t base = (size_t)(b * TT + t) * NDIM;
    const float gate = Z[base + HH + ch];
    const float hid  = Z[base + ch];
    const float proj = Z[base + 2 * HH + ch];
    const float c = sigf(-gate);
    const float v = sigf(gate) * gfun(hid);
    s = fmaf(c, s, v);
    const float gl = sigf(proj);
    const size_t hidx = (size_t)(b * TT + t) * HH + ch;
    const float hv = Hin[hidx];
    Hout[hidx] = fmaf(gl, s - hv, hv);  // gl*s + (1-gl)*hv
  }
}

// ---------------------------------------------------------------------------
extern "C" void kernel_launch(void* const* d_in, const int* in_sizes, int n_in,
                              void* d_out, int out_size, void* d_ws,
                              size_t ws_size, hipStream_t stream) {
  const float* h  = (const float*)d_in[0];
  const float* W0 = (const float*)d_in[1];
  const float* W1 = (const float*)d_in[2];
  float* out = (float*)d_out;

  float* z = (float*)d_ws;                         // [M, 3H] = 384 MB
  const size_t zElems = (size_t)MDIM * NDIM;
  float* cp    = z + zElems;                       // [B*NC, H]
  float* cs    = cp + (size_t)BB * NC * HH;
  float* carry = cs + (size_t)BB * NC * HH;

  const dim3 gGrid(MDIM / 128, NDIM / 64);
  const int scanBlocks = BB * NC * (HH / 256);     // 2048
  const int carryBlocks = (BB * HH) / 256;         // 32

  // ---- Layer 0: h -> out ----
  mingru_gemm_splitbf16<<<gGrid, 256, 0, stream>>>(h, W0, z);
  mingru_scan1<<<scanBlocks, 256, 0, stream>>>(z, cp, cs);
  mingru_scan2<<<carryBlocks, 256, 0, stream>>>(cp, cs, carry);
  mingru_scan3<<<scanBlocks, 256, 0, stream>>>(z, carry, h, out);

  // ---- Layer 1: out -> out (elementwise in-place across kernels is safe) ----
  mingru_gemm_splitbf16<<<gGrid, 256, 0, stream>>>(out, W1, z);
  mingru_scan1<<<scanBlocks, 256, 0, stream>>>(z, cp, cs);
  mingru_scan2<<<carryBlocks, 256, 0, stream>>>(cp, cs, carry);
  mingru_scan3<<<scanBlocks, 256, 0, stream>>>(z, carry, out, out);
}
